// SO_Module_9182640079333
// MI455X (gfx1250) — matrix operand order repaired and checked by reading, not compiled, not run on hardware
//
#include <hip/hip_runtime.h>
#include <hip/hip_bf16.h>
#include <math.h>

// ---------------------------------------------------------------------------
// Abbe litho simulation recast as f16 WMMA GEMMs for gfx1250 (wave32, WMMA).
//   M  = partial 2D DFT of mask (241x241 window, padded to 256)
//   H_s = M .* pupil_s            (per source)
//   G_s = H_s * V^T               (WMMA complex GEMM, K=256)
//   E_s = U * G_s                 (WMMA complex GEMM, K=256, async-LDS staged)
//   intensity += w_s * |E_s|^2
//   out: Is[d] = d^2*I/(weight*norm_I), Rs[d] = sigmoid(30*(Is-0.225))
// All twiddle factors precomputed as f16 planes in fragment-native layouts:
//   W[a][b]  = (cos, -sin)(2pi(a-120)b/512)   (K along b: M1-B, M2-A)
//   Vt[b][a] = (cos,  sin)(2pi(a-120)b/512)   (K along a: G-B, I-A/U)
// ---------------------------------------------------------------------------

typedef __attribute__((ext_vector_type(16))) _Float16 v16h;
typedef __attribute__((ext_vector_type(8)))  _Float16 v8h;
typedef __attribute__((ext_vector_type(8)))  float    v8f;

union Frag16 { v16h v; v8h h[2]; _Float16 e[16]; };
union Half8  { v8h  v; _Float16 e[8]; };

#define WMMA(a, b, c) \
  __builtin_amdgcn_wmma_f32_16x16x32_f16(false, (a), false, (b), (short)0, (c), false, false)

__device__ __forceinline__ float dev_pupil(float fg2) {
  float num = 1.0f - 0.11390625f * fg2;   // (M_RED*NA)^2
  float den = 1.0f - 0.87890625f * fg2;   // (NA/N_LIQUID)^2
  return sqrtf(sqrtf(num / den));
}

// wave32 lane decomposition: hs = lane/16, r = lane%16
// A-frag (16x32 f16):   element e -> K = hs*16 + e, row = r (kI: hs*8 + (e&7) + ((e>>3)<<4))
// B-frag (32x16 f16):   element e -> K = hs*16 + e, col = r (kI: same order as its A-frag)
// C/D (16x16 f32 v8f):  element e -> row M = hs*8 + e, col N = r

// ---------------------------------------------------------------------------
// Kernel W: source weights + normalization scalar
// ---------------------------------------------------------------------------
__global__ void litho_kW(const float* __restrict__ sp, const float* __restrict__ sfx,
                         const float* __restrict__ sfy, float* __restrict__ wbuf,
                         float* __restrict__ scal) {
  __shared__ float red[256];
  int t = threadIdx.x;
  float wv = 0.f, contrib = 0.f;
  if (t < 169) {
    float p   = sp[t];
    float val = 1.f / (1.f + __expf(-8.f * p));
    int valid = (val >= 0.001f);
    wv = valid ? val : 0.f;
    float fx = sfx[t], fy = sfy[t];
    float fg2 = valid ? (fx * fx + fy * fy) : 0.f;
    float obl = dev_pupil(fg2);
    contrib = wv * obl * obl;
  }
  wbuf[t] = wv;  // pads 169..255 with zero
  red[t] = contrib;
  __syncthreads();
  for (int s = 128; s > 0; s >>= 1) {
    if (t < s) red[t] += red[t + s];
    __syncthreads();
  }
  if (t == 0) scal[0] = 1.f / (1.44f * red[0]);  // 1/(weight*norm_I)
}

// ---------------------------------------------------------------------------
// Kernel T: precompute f16 twiddle planes
// ---------------------------------------------------------------------------
__global__ void litho_kT(_Float16* __restrict__ Wc, _Float16* __restrict__ Wsn,
                         _Float16* __restrict__ Vtc, _Float16* __restrict__ Vts) {
  int idx = blockIdx.x * 256 + threadIdx.x;   // 512 blocks -> 131072
  int a = idx >> 9, b = idx & 511;
  int m = ((a - 120) * b) & 511;
  float s, c;
  __sincosf((float)m * 0.012271846303085129f, &s, &c);  // 2*pi/512
  Wc [(size_t)a * 512 + b] = (_Float16)c;
  Wsn[(size_t)a * 512 + b] = (_Float16)(-s);
  Vtc[(size_t)b * 256 + a] = (_Float16)c;
  Vts[(size_t)b * 256 + a] = (_Float16)s;
}

// ---------------------------------------------------------------------------
// Kernel M1: Rt[i][y] = sum_x mask[y,x] * exp(-2pi*i*(i-120)*x/512)
// grid(8,4), out rows y (512), cols i (256), K = x (512)
// ---------------------------------------------------------------------------
__global__ void litho_kM1(const float* __restrict__ mask,
                          const _Float16* __restrict__ Wc, const _Float16* __restrict__ Wsn,
                          _Float16* __restrict__ RtR, _Float16* __restrict__ RtI) {
  int t = threadIdx.x;
  int w = t >> 5, lane = t & 31, hs = lane >> 4, r = lane & 15;
  int rt = w & 3, ct0 = (w >> 2) * 2;
  int y0 = blockIdx.x * 64 + rt * 16;
  int ib0 = blockIdx.y * 64;

  v8f zero = {};
  v8f accR[2], accI[2];
  accR[0] = accR[1] = zero;
  accI[0] = accI[1] = zero;

  const float* mrow = mask + (size_t)(y0 + r) * 512;
  for (int kk = 0; kk < 16; ++kk) {
    int x0 = kk * 32;
    Frag16 A;
#pragma unroll
    for (int e = 0; e < 16; ++e) {
      int k = hs * 16 + e;
      A.e[e] = (_Float16)mrow[x0 + k];
    }
#pragma unroll
    for (int c2 = 0; c2 < 2; ++c2) {
      int icol = ib0 + (ct0 + c2) * 16 + r;
      Frag16 Br, Bi;
      const _Float16* bc = Wc  + (size_t)icol * 512 + x0 + hs * 16;
      const _Float16* bs = Wsn + (size_t)icol * 512 + x0 + hs * 16;
      Br.h[0] = *(const v8h*)bc;
      Br.h[1] = *(const v8h*)(bc + 8);
      Bi.h[0] = *(const v8h*)bs;
      Bi.h[1] = *(const v8h*)(bs + 8);
      accR[c2] = WMMA(A.v, Br.v, accR[c2]);
      accI[c2] = WMMA(A.v, Bi.v, accI[c2]);
    }
  }
#pragma unroll
  for (int c2 = 0; c2 < 2; ++c2) {
    int icol = ib0 + (ct0 + c2) * 16 + r;
    Half8 pr, pi;
#pragma unroll
    for (int e = 0; e < 8; ++e) {
      pr.e[e] = (_Float16)accR[c2][e];
      pi.e[e] = (_Float16)accI[c2][e];
    }
    int yy = y0 + hs * 8;
    *(v8h*)(RtR + (size_t)icol * 512 + yy) = pr.v;
    *(v8h*)(RtI + (size_t)icol * 512 + yy) = pi.v;
  }
}

// ---------------------------------------------------------------------------
// Kernel M2: M[j][i] = (1/N^2) * sum_y exp(-2pi*i*(j-120)*y/512) * R[y][i]
// grid(4,4), out rows j (256), cols i (256), K = y (512)
// ---------------------------------------------------------------------------
__global__ void litho_kM2(const _Float16* __restrict__ RtR, const _Float16* __restrict__ RtI,
                          const _Float16* __restrict__ Wc, const _Float16* __restrict__ Wsn,
                          _Float16* __restrict__ Mr, _Float16* __restrict__ Mi) {
  int t = threadIdx.x;
  int w = t >> 5, lane = t & 31, hs = lane >> 4, r = lane & 15;
  int rt = w & 3, ct0 = (w >> 2) * 2;
  int j0 = blockIdx.x * 64 + rt * 16;
  int ib0 = blockIdx.y * 64;

  v8f zero = {};
  v8f accP[2], accQ[2], accM[2];
  accP[0] = accP[1] = zero;
  accQ[0] = accQ[1] = zero;
  accM[0] = accM[1] = zero;

  int j = j0 + r;
  const _Float16* ac = Wc  + (size_t)j * 512;
  const _Float16* as = Wsn + (size_t)j * 512;
  for (int kk = 0; kk < 16; ++kk) {
    int y0k = kk * 32;
    Frag16 Ar, Ai;
    Ar.h[0] = *(const v8h*)(ac + y0k + hs * 16);
    Ar.h[1] = *(const v8h*)(ac + y0k + hs * 16 + 8);
    Ai.h[0] = *(const v8h*)(as + y0k + hs * 16);
    Ai.h[1] = *(const v8h*)(as + y0k + hs * 16 + 8);
#pragma unroll
    for (int c2 = 0; c2 < 2; ++c2) {
      int icol = ib0 + (ct0 + c2) * 16 + r;
      Frag16 Br, Bi;
      const _Float16* br = RtR + (size_t)icol * 512 + y0k + hs * 16;
      Br.h[0] = *(const v8h*)br;
      Br.h[1] = *(const v8h*)(br + 8);
      const _Float16* bi = RtI + (size_t)icol * 512 + y0k + hs * 16;
      Bi.h[0] = *(const v8h*)bi;
      Bi.h[1] = *(const v8h*)(bi + 8);
      accP[c2] = WMMA(Ar.v, Br.v, accP[c2]);
      accQ[c2] = WMMA(Ai.v, Bi.v, accQ[c2]);
      accM[c2] = WMMA(Ar.v, Bi.v, accM[c2]);
      accM[c2] = WMMA(Ai.v, Br.v, accM[c2]);
    }
  }
  const float SCALE = 1.f / 262144.f;  // 1/(512*512)
#pragma unroll
  for (int c2 = 0; c2 < 2; ++c2) {
    int icol = ib0 + (ct0 + c2) * 16 + r;
#pragma unroll
    for (int e = 0; e < 8; ++e) {
      int jj = j0 + hs * 8 + e;
      Mr[(size_t)jj * 256 + icol] = (_Float16)((accP[c2][e] - accQ[c2][e]) * SCALE);
      Mi[(size_t)jj * 256 + icol] = (_Float16)(accM[c2][e] * SCALE);
    }
  }
}

// ---------------------------------------------------------------------------
// Kernel P: H_s = M .* pupil_s (f16), zero on padding / outside rho2<=1
// ---------------------------------------------------------------------------
__global__ void litho_kP(const _Float16* __restrict__ Mr, const _Float16* __restrict__ Mi,
                         const float* __restrict__ sfx, const float* __restrict__ sfy,
                         int s0, _Float16* __restrict__ Hh) {
  int ls = blockIdx.x;
  float fx = sfx[s0 + ls], fy = sfy[s0 + ls];
  _Float16* Hr = Hh + (size_t)ls * 131072;
  _Float16* Hi = Hr + 65536;
  int base = blockIdx.y * 1024 + threadIdx.x * 4;
  int jj = base >> 8;
  int i0 = base & 255;
  float Gv = 0.01f * (float)(jj - 120) + fy;
  float g2 = Gv * Gv;
#pragma unroll
  for (int q = 0; q < 4; ++q) {
    int ii = i0 + q;
    float Fv = 0.01f * (float)(ii - 120) + fx;
    float rho2 = Fv * Fv + g2;
    float p = (rho2 <= 1.0f && jj <= 240 && ii <= 240) ? dev_pupil(rho2) : 0.f;
    size_t idx = (size_t)jj * 256 + ii;
    Hr[idx] = (_Float16)((float)Mr[idx] * p);
    Hi[idx] = (_Float16)((float)Mi[idx] * p);
  }
}

// ---------------------------------------------------------------------------
// Kernel G: Gt_s[x][j] = sum_i H_s[j][i] * exp(+2pi*i*(i-120)*x/512)
// grid(cnt, 8 x-tiles of 64), out 256(j) x 64(x), K = i (256)
// ---------------------------------------------------------------------------
__global__ void litho_kG(const _Float16* __restrict__ Hh,
                         const _Float16* __restrict__ Vtc, const _Float16* __restrict__ Vts,
                         _Float16* __restrict__ Gh) {
  int t = threadIdx.x;
  int w = t >> 5, lane = t & 31, hs = lane >> 4, r = lane & 15;
  int ls = blockIdx.x;
  int xb = blockIdx.y * 64;
  const _Float16* Hr = Hh + (size_t)ls * 131072;
  const _Float16* Hi = Hr + 65536;
  _Float16* GtR = Gh + (size_t)ls * 262144;
  _Float16* GtI = GtR + 131072;

  v8f zero = {};
  v8f accP[8], accQ[8], accE[8];
#pragma unroll
  for (int u = 0; u < 8; ++u) { accP[u] = zero; accQ[u] = zero; accE[u] = zero; }

  for (int kk = 0; kk < 8; ++kk) {
    int i0 = kk * 32;
    Frag16 ArF[2], AiF[2];
#pragma unroll
    for (int a2 = 0; a2 < 2; ++a2) {
      int jj = (w + a2 * 8) * 16 + r;
      const _Float16* p0 = Hr + (size_t)jj * 256 + i0 + hs * 16;
      ArF[a2].h[0] = *(const v8h*)p0;
      ArF[a2].h[1] = *(const v8h*)(p0 + 8);
      const _Float16* p1 = Hi + (size_t)jj * 256 + i0 + hs * 16;
      AiF[a2].h[0] = *(const v8h*)p1;
      AiF[a2].h[1] = *(const v8h*)(p1 + 8);
    }
#pragma unroll
    for (int ct = 0; ct < 4; ++ct) {
      int x = xb + ct * 16 + r;
      Frag16 Vr, Vi;
      const _Float16* vc = Vtc + (size_t)x * 256 + i0 + hs * 16;
      const _Float16* vs = Vts + (size_t)x * 256 + i0 + hs * 16;
      Vr.h[0] = *(const v8h*)vc;
      Vr.h[1] = *(const v8h*)(vc + 8);
      Vi.h[0] = *(const v8h*)vs;
      Vi.h[1] = *(const v8h*)(vs + 8);
#pragma unroll
      for (int a2 = 0; a2 < 2; ++a2) {
        int u = a2 * 4 + ct;
        accP[u] = WMMA(ArF[a2].v, Vr.v, accP[u]);
        accQ[u] = WMMA(AiF[a2].v, Vi.v, accQ[u]);
        accE[u] = WMMA(ArF[a2].v, Vi.v, accE[u]);
        accE[u] = WMMA(AiF[a2].v, Vr.v, accE[u]);
      }
    }
  }
#pragma unroll
  for (int a2 = 0; a2 < 2; ++a2) {
    int jb = (w + a2 * 8) * 16 + hs * 8;
#pragma unroll
    for (int ct = 0; ct < 4; ++ct) {
      int u = a2 * 4 + ct;
      int x = xb + ct * 16 + r;
      Half8 gr, gi;
#pragma unroll
      for (int e = 0; e < 8; ++e) {
        gr.e[e] = (_Float16)(accP[u][e] - accQ[u][e]);
        gi.e[e] = (_Float16)accE[u][e];
      }
      *(v8h*)(GtR + (size_t)x * 256 + jb) = gr.v;
      *(v8h*)(GtI + (size_t)x * 256 + jb) = gi.v;
    }
  }
}

// ---------------------------------------------------------------------------
// Kernel I: E tiles = U * G_s; intensity += w_s * |E|^2
// grid(16,8): 64(y) x 32(x) tiles. G slice double-buffered in LDS via
// global_load_async_to_lds_b128 (ASYNCcnt) with 16B XOR swizzle.
// ---------------------------------------------------------------------------
__global__ void litho_kI(const _Float16* __restrict__ Gh, const float* __restrict__ wbuf,
                         const _Float16* __restrict__ Vtc, const _Float16* __restrict__ Vts,
                         int s0, int cnt, int accumulate, float* __restrict__ inten) {
  __shared__ __align__(16) _Float16 Gs[2][16384];  // 2 x 32KB (Gr 8192 | Gi 8192)
  int t = threadIdx.x, w = t >> 5, lane = t & 31, hs = lane >> 4, r = lane & 15;
  int rt = w & 3, ct = w >> 2;  // 4 row-tiles x 2 col-tiles
  int yb = blockIdx.y * 64, xb = blockIdx.x * 32;

  // cache U fragments (source-independent) in registers, loaded from Vt planes
  v16h Ur[8], Ui[8];
  {
    int y = yb + rt * 16 + r;
    const _Float16* uc = Vtc + (size_t)y * 256;
    const _Float16* us = Vts + (size_t)y * 256;
#pragma unroll
    for (int kk = 0; kk < 8; ++kk) {
      Frag16 a, b;
      a.h[0] = *(const v8h*)(uc + kk * 32 + hs * 8);
      a.h[1] = *(const v8h*)(uc + kk * 32 + hs * 8 + 16);
      b.h[0] = *(const v8h*)(us + kk * 32 + hs * 8);
      b.h[1] = *(const v8h*)(us + kk * 32 + hs * 8 + 16);
      Ur[kk] = a.v;
      Ui[kk] = b.v;
    }
  }

  // async stage of one source's G slice (32 x-rows, both planes) into buffer
  auto issue = [&](int ls, int buf) {
    const char* bR = (const char*)(Gh + (size_t)ls * 262144 + (size_t)xb * 256);
    const char* bI = (const char*)(Gh + (size_t)ls * 262144 + 131072 + (size_t)xb * 256);
    _Float16* dR = &Gs[buf][0];
    _Float16* dI = &Gs[buf][8192];
    for (int c = t; c < 1024; c += 256) {       // 1024 x 16B per plane
      int x = c >> 5, cc = c & 31;
      int d16 = (x << 5) + (cc ^ (x & 15));     // XOR-swizzled 16B chunk
      uint32_t laR = (uint32_t)(uintptr_t)(dR + d16 * 8);
      uint32_t laI = (uint32_t)(uintptr_t)(dI + d16 * 8);
      asm volatile("global_load_async_to_lds_b128 %0, %1, off"
                   :: "v"(laR), "v"((unsigned long long)(uintptr_t)(bR + c * 16)) : "memory");
      asm volatile("global_load_async_to_lds_b128 %0, %1, off"
                   :: "v"(laI), "v"((unsigned long long)(uintptr_t)(bI + c * 16)) : "memory");
    }
  };

  v8f zero = {};
  v8f iacc = zero;

  int buf = 0;
  issue(0, 0);
  for (int ls = 0; ls < cnt; ++ls) {
    asm volatile("s_wait_asynccnt 0" ::: "memory");
    __syncthreads();
    if (ls + 1 < cnt) issue(ls + 1, buf ^ 1);  // prefetch next source

    float wv = wbuf[s0 + ls];
    if (wv > 0.f) {
      const _Float16* gR = &Gs[buf][0];
      const _Float16* gI = &Gs[buf][8192];
      v8f accP = zero, accQ = zero, accE = zero;
      int xl = ct * 16 + r;
#pragma unroll
      for (int kk = 0; kk < 8; ++kk) {
        int cc0 = kk * 4 + hs;  // 16B chunk index within row
        const _Float16* b0 = gR + xl * 256 + ((cc0 ^ (xl & 15)) << 3);
        const _Float16* b1 = gR + xl * 256 + (((cc0 + 2) ^ (xl & 15)) << 3);
        Frag16 Br, Bi;
        Br.h[0] = *(const v8h*)b0;
        Br.h[1] = *(const v8h*)b1;
        Bi.h[0] = *(const v8h*)(b0 + 8192);
        Bi.h[1] = *(const v8h*)(b1 + 8192);
        accP = WMMA(Ur[kk], Br.v, accP);
        accQ = WMMA(Ui[kk], Bi.v, accQ);
        accE = WMMA(Ur[kk], Bi.v, accE);
        accE = WMMA(Ui[kk], Br.v, accE);
      }
#pragma unroll
      for (int e = 0; e < 8; ++e) {
        float er = accP[e] - accQ[e];
        float ei = accE[e];
        iacc[e] += wv * (er * er + ei * ei);
      }
    }
    __syncthreads();
    buf ^= 1;
  }

#pragma unroll
  for (int e = 0; e < 8; ++e) {
    int y = yb + rt * 16 + hs * 8 + e;
    int x = xb + ct * 16 + r;
    size_t idx = (size_t)y * 512 + x;
    float v = iacc[e];
    if (accumulate) v += inten[idx];
    inten[idx] = v;
  }
}

// ---------------------------------------------------------------------------
// Kernel F: doses + resist sigmoid -> out (2,3,512,512)
// ---------------------------------------------------------------------------
__global__ void litho_kF(const float* __restrict__ inten, const float* __restrict__ scal,
                         float* __restrict__ out) {
  int p = blockIdx.x * 256 + threadIdx.x;
  float inv = scal[0];
  float base = inten[p] * inv;
  const float d2[3] = {0.98f * 0.98f, 1.0f, 1.02f * 1.02f};
#pragma unroll
  for (int di = 0; di < 3; ++di) {
    float I = d2[di] * base;
    out[(size_t)di * 262144 + p] = I;
    out[(size_t)(3 + di) * 262144 + p] = 1.f / (1.f + __expf(-30.f * (I - 0.225f)));
  }
}

// ---------------------------------------------------------------------------
extern "C" void kernel_launch(void* const* d_in, const int* in_sizes, int n_in,
                              void* d_out, int out_size, void* d_ws, size_t ws_size,
                              hipStream_t stream) {
  const float* mask = (const float*)d_in[0];   // (512,512)
  const float* sprm = (const float*)d_in[1];   // (13,13)
  const float* sfx  = (const float*)d_in[2];   // (169,)
  const float* sfy  = (const float*)d_in[3];   // (169,)
  float* out = (float*)d_out;
  char* p = (char*)d_ws;
  char* ws0 = p;

  float*    wbuf = (float*)p;    p += 1024;
  float*    scal = (float*)p;    p += 256;
  _Float16* Wc   = (_Float16*)p; p += 262144;   // [256][512] cos
  _Float16* Wsn  = (_Float16*)p; p += 262144;   // [256][512] -sin
  _Float16* Vtc  = (_Float16*)p; p += 262144;   // [512][256] cos
  _Float16* Vts  = (_Float16*)p; p += 262144;   // [512][256] sin
  _Float16* RtR  = (_Float16*)p; p += 262144;
  _Float16* RtI  = (_Float16*)p; p += 262144;
  _Float16* Mr   = (_Float16*)p; p += 131072;
  _Float16* Mi   = (_Float16*)p; p += 131072;
  float*    intn = (float*)p;    p += 1048576;
  size_t base = (size_t)(p - ws0);

  const size_t per_src = 786432;  // H (256KB) + G (512KB)
  int SC = 1;
  if (ws_size > base + per_src) SC = (int)((ws_size - base) / per_src);
  if (SC > 169) SC = 169;
  if (SC < 1) SC = 1;
  _Float16* Hh = (_Float16*)p;
  _Float16* Gh = (_Float16*)(p + (size_t)SC * 262144);

  litho_kW<<<1, 256, 0, stream>>>(sprm, sfx, sfy, wbuf, scal);
  litho_kT<<<512, 256, 0, stream>>>(Wc, Wsn, Vtc, Vts);
  litho_kM1<<<dim3(8, 4), 256, 0, stream>>>(mask, Wc, Wsn, RtR, RtI);
  litho_kM2<<<dim3(4, 4), 256, 0, stream>>>(RtR, RtI, Wc, Wsn, Mr, Mi);

  int chunk = 0;
  for (int s0 = 0; s0 < 169; s0 += SC, ++chunk) {
    int cnt = 169 - s0;
    if (cnt > SC) cnt = SC;
    litho_kP<<<dim3(cnt, 64), 256, 0, stream>>>(Mr, Mi, sfx, sfy, s0, Hh);
    litho_kG<<<dim3(cnt, 8), 256, 0, stream>>>(Hh, Vtc, Vts, Gh);
    litho_kI<<<dim3(16, 8), 256, 0, stream>>>(Gh, wbuf, Vtc, Vts, s0, cnt,
                                              chunk == 0 ? 0 : 1, intn);
  }
  litho_kF<<<1024, 256, 0, stream>>>(intn, scal, out);
}
